// TwoWayTransformer_36919538876940
// MI455X (gfx1250) — matrix operand order repaired and checked by reading, not compiled, not run on hardware
//
#include <hip/hip_runtime.h>
#include <math.h>

// ============================================================================
// TwoWayTransformer forward for MI455X (gfx1250, wave32, WMMA).
// All GEMMs + attention inner products use v_wmma_f32_16x16x32_f16
// (f16 operands, f32 accumulation). Attention is flash-style: one wave per
// 16-query tile streaming 32-key tiles, online softmax in f32.
//
// Round-3 changes: flash attention K/V/Q/mask tiles are staged with the CDNA5
// async DMA path (global_load_async_to_lds_b128 / _b32, ASYNCcnt) into
// double-buffered per-wave LDS slices -- no VGPR round trip, next tile's DMA
// overlaps current tile's WMMAs.  f32->f16 happens in the fragment build via
// v_cvt_pk_rtz_f16_f32 on paired loads.  GEMM unchanged (its inner loop is
// already ds_load_b128 -> 4x v_wmma); prefetch switched to near-cache
// locality and made branchless.
// ============================================================================

typedef __attribute__((ext_vector_type(16))) _Float16 v16h;
typedef __attribute__((ext_vector_type(8)))  float    v8f;

union AFrag { v16h v; _Float16 e[16]; unsigned u[8]; };
union CFrag { v8f  v; float    e[8];  };

#define WMMA_F32_F16(A, Bm, C) \
  __builtin_amdgcn_wmma_f32_16x16x32_f16(false, (A), false, (Bm), (short)0, (C), false, false)

// Zero-cost fence: keeps the compiler from reordering LDS ops across it.
#define LDS_FENCE() asm volatile("" ::: "memory")

// CDNA5 async global->LDS DMA (CDNA5 ISA; inline asm per
// bridge-doc guidance since the clang builtin arity differs across toolchains).
// %0 = LDS byte address (VGPR), %1 = 64-bit global address (VGPR pair).
#define ASYNC_LDS_B128(ldsoff, gptr) \
  asm volatile("global_load_async_to_lds_b128 %0, %1, off" :: "v"(ldsoff), "v"(gptr) : "memory")
#define ASYNC_LDS_B32(ldsoff, gptr) \
  asm volatile("global_load_async_to_lds_b32 %0, %1, off" :: "v"(ldsoff), "v"(gptr) : "memory")
#define WAIT_ASYNC(n) asm volatile("s_wait_asynccnt %0" :: "i"(n) : "memory")
#define WAIT_DS0()    asm volatile("s_wait_dscnt 0x0" ::: "memory")

// pack two f32 -> one VGPR of two f16 (v_cvt_pk_rtz_f16_f32)
static __device__ inline unsigned pkh(float x, float y) {
  auto r = __builtin_amdgcn_cvt_pkrtz(x, y);
  unsigned u; __builtin_memcpy(&u, &r, 4); return u;
}

namespace cfg {
constexpr int B = 8, NQ = 256, KK = 4096, ED = 256, NH = 8, IDIM = 128, MLPD = 2048;
}

// ---------------------------------------------------------------------------
// Sinusoidal positional-encoding gather: pe[b,k,d] = sin/cos(pos * freq(d))
// ---------------------------------------------------------------------------
__global__ void pe_kernel(const long long* __restrict__ pos, float* __restrict__ pe, int n) {
  int i = blockIdx.x * 256 + threadIdx.x;
  if (i >= n) return;
  int d = i & 255;                       // ED == 256
  float p = (float)pos[i >> 8];
  float freq = __expf(-(float)(d & ~1) * (9.210340371976184f / 256.f));
  float ang = p * freq;
  pe[i] = (d & 1) ? __cosf(ang) : __sinf(ang);
}

__global__ void add_kernel(const float* __restrict__ a, const float* __restrict__ b,
                           float* __restrict__ c, int n) {
  int i = blockIdx.x * 256 + threadIdx.x;
  if (i < n) c[i] = a[i] + b[i];
}

// ---------------------------------------------------------------------------
// GEMM: C[M,N] = act(A[M,KD] @ W[KD,N] + bias).  M,N % 64 == 0, KD % 32 == 0.
// Block = 128 threads = 4 waves; block tile 64x64; each wave a 32x32 subtile
// (2 A-frags x 2 B-frags -> 4 WMMAs per 32-deep K step).  A staged row-major,
// W staged *transposed* (sBt[n][k]) so B-fragment K-pairs are contiguous and
// every fragment VGPR fills with one 32-bit ds load (merged to ds_load_b128).
// ---------------------------------------------------------------------------
template<bool RELU>
__global__ __launch_bounds__(128)
void gemm_kernel(const float* __restrict__ A, const float* __restrict__ W,
                 const float* __restrict__ bias, float* __restrict__ C,
                 int M, int N, int KD) {
  __shared__ alignas(16) _Float16 sA [64][40];
  __shared__ alignas(16) _Float16 sBt[64][40];   // transposed: [n][k]
  const int t = threadIdx.x;
  const int wv = t >> 5, lane = t & 31, half = lane >> 4, l16 = lane & 15;
  const int mb = blockIdx.y * 64, nb = blockIdx.x * 64;
  const int wm = (wv >> 1) * 32, wn = (wv & 1) * 32;
  const int arow = t >> 1, acg = (t & 1) * 16;   // A staging: 64 rows x 16 cols
  const int brow = t >> 2, bcg = (t & 3) * 16;   // W staging: 32 k   x 16 cols

  CFrag a00, a01, a10, a11;
#pragma unroll
  for (int i = 0; i < 8; ++i) { a00.e[i] = 0.f; a01.e[i] = 0.f; a10.e[i] = 0.f; a11.e[i] = 0.f; }

  for (int kb = 0; kb < KD; kb += 32) {
    const float* ap = A + (size_t)(mb + arow) * KD + kb + acg;
    const float* wp = W + (size_t)(kb + brow) * N + nb + bcg;
    {   // branchless near-cache prefetch of the next K tiles
      const int nxt = (kb + 32 < KD) ? 32 : 0;
      __builtin_prefetch(ap + nxt, 0, 3);
      __builtin_prefetch(wp + (size_t)nxt * N, 0, 3);
    }
#pragma unroll
    for (int i = 0; i < 16; i += 4) {
      float4 f = *(const float4*)(ap + i);
      sA[arow][acg + i + 0] = (_Float16)f.x; sA[arow][acg + i + 1] = (_Float16)f.y;
      sA[arow][acg + i + 2] = (_Float16)f.z; sA[arow][acg + i + 3] = (_Float16)f.w;
    }
#pragma unroll
    for (int i = 0; i < 16; i += 4) {
      float4 f = *(const float4*)(wp + i);
      sBt[bcg + i + 0][brow] = (_Float16)f.x; sBt[bcg + i + 1][brow] = (_Float16)f.y;
      sBt[bcg + i + 2][brow] = (_Float16)f.z; sBt[bcg + i + 3][brow] = (_Float16)f.w;
    }
    __syncthreads();

    AFrag af0, af1, bf0, bf1;
    const unsigned* pa0 = (const unsigned*)&sA [wm + l16     ][0];
    const unsigned* pa1 = (const unsigned*)&sA [wm + 16 + l16][0];
    const unsigned* pb0 = (const unsigned*)&sBt[wn + l16     ][0];
    const unsigned* pb1 = (const unsigned*)&sBt[wn + 16 + l16][0];
#pragma unroll
    for (int v = 0; v < 8; ++v) {
      const int ka = (v & 3) + half * 4 + (v >> 2) * 8;   // A-frag K-pair index
      const int kbp = (v & 3) + half * 4 + (v >> 2) * 8;  // B-frag K-pair index
      af0.u[v] = pa0[ka];  af1.u[v] = pa1[ka];
      bf0.u[v] = pb0[kbp]; bf1.u[v] = pb1[kbp];
    }
    a00.v = WMMA_F32_F16(af0.v, bf0.v, a00.v);
    a01.v = WMMA_F32_F16(af0.v, bf1.v, a01.v);
    a10.v = WMMA_F32_F16(af1.v, bf0.v, a10.v);
    a11.v = WMMA_F32_F16(af1.v, bf1.v, a11.v);
    __syncthreads();
  }

  const float b0 = bias[nb + wn + l16];
  const float b1 = bias[nb + wn + 16 + l16];
#pragma unroll
  for (int v = 0; v < 8; ++v) {
    const int r0 = mb + wm + v + 8 * half;
    const int r1 = r0 + 16;
    const int c0 = nb + wn + l16, c1 = c0 + 16;
    float x;
    x = a00.e[v] + b0; if (RELU) x = fmaxf(x, 0.f); C[(size_t)r0 * N + c0] = x;
    x = a01.e[v] + b1; if (RELU) x = fmaxf(x, 0.f); C[(size_t)r0 * N + c1] = x;
    x = a10.e[v] + b0; if (RELU) x = fmaxf(x, 0.f); C[(size_t)r1 * N + c0] = x;
    x = a11.e[v] + b1; if (RELU) x = fmaxf(x, 0.f); C[(size_t)r1 * N + c1] = x;
  }
}

// ---------------------------------------------------------------------------
// Flash attention: one wave per (b, head, 16-query tile). DH = 16 or 32
// (zero-padded to K=32 for the S = Q.K^T WMMA).  Streams keys 32 at a time.
// K/V/Q/mask tiles staged f32 by async DMA (double-buffered, ASYNCcnt);
// f32->f16 conversion fused into fragment builds with v_cvt_pk_rtz_f16_f32.
// Per-wave LDS slices; no block barriers (LDS is in-order per wave).
// ---------------------------------------------------------------------------
template<int DH, bool MASKED>
__global__ __launch_bounds__(128)
void flash_attn_kernel(const float* __restrict__ Q, const float* __restrict__ Km,
                       const float* __restrict__ Vm, const int* __restrict__ mask,
                       float* __restrict__ O, int Nq, int Nk, int width, float scale) {
  __shared__ alignas(16) float    sQf[4][16][36];      // [q][dim],  dim padded to 32
  __shared__ alignas(16) float    sKf[4][2][32][36];   // [key][dim] double-buffered
  __shared__ alignas(16) float    sVf[4][2][32][36];   // [key][dim] double-buffered
  __shared__ alignas(16) _Float16 sP [4][16][36];      // [q][key-in-tile]
  __shared__ alignas(16) int      sMI[4][2][32];

  const int wv = threadIdx.x >> 5;
  const int lane = threadIdx.x & 31;
  const int half = lane >> 4, l16 = lane & 15;
  const int gw = blockIdx.x * 4 + wv;
  const int qtiles = Nq >> 4;
  const int b  = gw / (cfg::NH * qtiles);
  const int rm = gw % (cfg::NH * qtiles);
  const int h  = rm / qtiles;
  const int qb = (rm % qtiles) * 16;

  constexpr int TOPS = 2 * (DH / 4) + (MASKED ? 1 : 0);  // async ops per tile

  // Zero the GEMM-K padding once (DH==16 only); async never touches it.
  if (DH < 32) {
#pragma unroll
    for (int c = DH; c < 32; ++c) {
      sKf[wv][0][lane][c] = 0.f;
      sKf[wv][1][lane][c] = 0.f;
      if (lane < 16) sQf[wv][lane][c] = 0.f;
    }
  }

  // Async-stage the Q tile (one row per lane, lanes 0..15).
  if (lane < 16) {
    const float* qr = Q + ((size_t)b * Nq + qb + lane) * width + h * DH;
    const unsigned lq = (unsigned)(size_t)&sQf[wv][lane][0];
#pragma unroll
    for (int c = 0; c < DH; c += 4) ASYNC_LDS_B128(lq + c * 4, qr + c);
  }

  auto issue_tile = [&](int kb2, int buf) {
    const int kr = kb2 + lane;                     // one key row per lane
    const float* kp = Km + ((size_t)b * Nk + kr) * width + h * DH;
    const float* vp = Vm + ((size_t)b * Nk + kr) * width + h * DH;
    const unsigned lk = (unsigned)(size_t)&sKf[wv][buf][lane][0];
    const unsigned lv = (unsigned)(size_t)&sVf[wv][buf][lane][0];
#pragma unroll
    for (int c = 0; c < DH; c += 4) {
      ASYNC_LDS_B128(lk + c * 4, kp + c);
      ASYNC_LDS_B128(lv + c * 4, vp + c);
    }
    if (MASKED)
      ASYNC_LDS_B32((unsigned)(size_t)&sMI[wv][buf][lane], mask + (size_t)b * Nk + kr);
  };

  issue_tile(0, 0);
  WAIT_ASYNC(TOPS);            // Q tile resident (tile 0 may still be in flight)

  AFrag aQ;
  {
    const float2* pq = (const float2*)&sQf[wv][l16][0];
#pragma unroll
    for (int v = 0; v < 8; ++v) {
      const float2 d = pq[(v & 3) + half * 4 + (v >> 2) * 8];
      aQ.u[v] = pkh(d.x, d.y);
    }
  }

  CFrag o0, o1;
  float m[8], l[8];
#pragma unroll
  for (int v = 0; v < 8; ++v) { o0.e[v] = 0.f; o1.e[v] = 0.f; m[v] = -1e30f; l[v] = 0.f; }

  const int ntiles = Nk >> 5;
  for (int kt = 0; kt < ntiles; ++kt) {
    const int buf = kt & 1;
    if (kt + 1 < ntiles) {
      WAIT_DS0();                       // finish reads of buf^1 before DMA overwrites it
      issue_tile((kt + 1) * 32, buf ^ 1);
      WAIT_ASYNC(TOPS);                 // current tile resident; next tile in flight
    } else {
      WAIT_ASYNC(0);
    }

    // S tile = Q (16xDH) . K^T (DHx32) -> two 16x16 f32 frags
    AFrag bK0, bK1;
    {
      const float2* pk0 = (const float2*)&sKf[wv][buf][l16     ][0];
      const float2* pk1 = (const float2*)&sKf[wv][buf][16 + l16][0];
#pragma unroll
      for (int v = 0; v < 8; ++v) {
        const int kd = (v & 3) + half * 4 + (v >> 2) * 8;  // dim-pair index
        const float2 d0 = pk0[kd], d1 = pk1[kd];
        bK0.u[v] = pkh(d0.x, d0.y);
        bK1.u[v] = pkh(d1.x, d1.y);
      }
    }
    CFrag s0, s1, z;
#pragma unroll
    for (int v = 0; v < 8; ++v) z.e[v] = 0.f;
    s0.v = WMMA_F32_F16(aQ.v, bK0.v, z.v);
    s1.v = WMMA_F32_F16(aQ.v, bK1.v, z.v);

    const float add0 = MASKED ? (sMI[wv][buf][l16]      ? 0.f : -1e9f) : 0.f;
    const float add1 = MASKED ? (sMI[wv][buf][16 + l16] ? 0.f : -1e9f) : 0.f;
#pragma unroll
    for (int v = 0; v < 8; ++v) {
      s0.e[v] = s0.e[v] * scale + add0;
      s1.e[v] = s1.e[v] * scale + add1;
      float t = fmaxf(s0.e[v], s1.e[v]);
#pragma unroll
      for (int off = 8; off > 0; off >>= 1) t = fmaxf(t, __shfl_xor(t, off, 16));
      const float mn = fmaxf(m[v], t);
      const float al = __expf(m[v] - mn);
      const float p0 = __expf(s0.e[v] - mn);
      const float p1 = __expf(s1.e[v] - mn);
      float rs = p0 + p1;
#pragma unroll
      for (int off = 8; off > 0; off >>= 1) rs += __shfl_xor(rs, off, 16);
      l[v] = l[v] * al + rs;
      m[v] = mn;
      o0.e[v] *= al;
      if (DH == 32) o1.e[v] *= al;
      sP[wv][v + 8 * half][l16]      = (_Float16)p0;   // repack P via LDS
      sP[wv][v + 8 * half][16 + l16] = (_Float16)p1;
    }
    LDS_FENCE();

    // O += P (16x32) . V (32xDH)
    AFrag aP, bV0, bV1;
    {
      const unsigned* pp = (const unsigned*)&sP[wv][l16][0];
      const float (*vt)[36] = sVf[wv][buf];
#pragma unroll
      for (int v = 0; v < 8; ++v) {
        aP.u[v] = pp[(v & 3) + half * 4 + (v >> 2) * 8];
        const int kd = 2 * ((v & 3) + half * 4 + (v >> 2) * 8);  // key-pair within tile
        bV0.u[v] = pkh(vt[kd][l16], vt[kd + 1][l16]);
        if (DH == 32) bV1.u[v] = pkh(vt[kd][16 + l16], vt[kd + 1][16 + l16]);
      }
    }
    o0.v = WMMA_F32_F16(aP.v, bV0.v, o0.v);
    if (DH == 32) o1.v = WMMA_F32_F16(aP.v, bV1.v, o1.v);
    LDS_FENCE();
  }

#pragma unroll
  for (int v = 0; v < 8; ++v) {
    const float inv = 1.f / l[v];
    float* orow = O + ((size_t)b * Nq + qb + v + 8 * half) * width + h * DH;
    orow[l16] = o0.e[v] * inv;
    if (DH == 32) orow[16 + l16] = o1.e[v] * inv;
  }
}

// ---------------------------------------------------------------------------
// LayerNorm over last dim (256), optional fused residual:  out = LN(x + res)
// ---------------------------------------------------------------------------
__global__ __launch_bounds__(256)
void layernorm_kernel(const float* __restrict__ x, const float* __restrict__ res,
                      const float* __restrict__ g, const float* __restrict__ bta,
                      float* __restrict__ out) {
  __shared__ float red[256];
  const int row = blockIdx.x, t = threadIdx.x;
  const size_t idx = (size_t)row * 256 + t;
  float v = x[idx] + (res ? res[idx] : 0.f);
  red[t] = v; __syncthreads();
#pragma unroll
  for (int s = 128; s > 0; s >>= 1) { if (t < s) red[t] += red[t + s]; __syncthreads(); }
  const float mean = red[0] * (1.f / 256.f);
  __syncthreads();
  const float d = v - mean;
  red[t] = d * d; __syncthreads();
#pragma unroll
  for (int s = 128; s > 0; s >>= 1) { if (t < s) red[t] += red[t + s]; __syncthreads(); }
  const float var = red[0] * (1.f / 256.f);
  out[idx] = d * rsqrtf(var + 1e-5f) * g[t] + bta[t];
}

// ===========================================================================
// Host orchestration
// ===========================================================================
extern "C" void kernel_launch(void* const* d_in, const int* in_sizes, int n_in,
                              void* d_out, int out_size, void* d_ws, size_t ws_size,
                              hipStream_t stream) {
  using namespace cfg;
  (void)in_sizes; (void)n_in; (void)out_size; (void)ws_size;

  const float*     vlm   = (const float*)d_in[0];     // [B,K,ED]
  const long long* pos   = (const long long*)d_in[1]; // [B,K] int64
  const float*     seg   = (const float*)d_in[2];     // [B,NQ,ED]  (== query_pe)
  const int*       amask = (const int*)d_in[3];       // [B,K] int32

  // ---- params: jax pytree-flatten order (dict keys sorted) ----
  int p = 4;
  auto F = [&]() { return (const float*)d_in[p++]; };
  struct Lin   { const float *w, *b; };
  struct AttnP { Lin q, k, v, o; };
  struct LNP   { const float *g, *b; };
  auto rdLin  = [&](Lin& Lx)  { Lx.b = F(); Lx.w = F(); };                 // {'b','w'}
  auto rdAttn = [&](AttnP& A) { rdLin(A.k); rdLin(A.o); rdLin(A.q); rdLin(A.v); }; // k,o,q,v
  auto rdLN   = [&](LNP& N)   { N.b = F(); N.g = F(); };                   // {'b','g'}
  struct LayerP { AttnP self_attn, cross_t2i, cross_i2t; Lin lin1, lin2; LNP n1, n2, n3, n4; };

  AttnP finalA; rdAttn(finalA);              // 'final'
  LayerP L[2];                               // 'layers'
  for (int i = 0; i < 2; ++i) {              // keys: cross_i2t, cross_t2i, mlp, norm1..4, self_attn
    rdAttn(L[i].cross_i2t);
    rdAttn(L[i].cross_t2i);
    rdLin(L[i].lin1); rdLin(L[i].lin2);
    rdLN(L[i].n1); rdLN(L[i].n2); rdLN(L[i].n3); rdLN(L[i].n4);
    rdAttn(L[i].self_attn);
  }
  LNP nF; rdLN(nF);                          // 'norm_final'

  // ---- workspace (~225 MB of fp32) ----
  float* ws = (float*)d_ws;
  size_t off = 0;
  auto alloc = [&](size_t n) { float* q = ws + off; off += n; return q; };
  const size_t Kel = (size_t)B * KK * ED;    // 8.39M
  const size_t Qel = (size_t)B * NQ * ED;    // 0.52M
  const size_t KI  = (size_t)B * KK * IDIM;  // 4.19M
  float* PE   = alloc(Kel);
  float* qry  = alloc(Qel);
  float* keys = alloc(Kel);
  float* qtmp = alloc(Qel);
  float* ktmp = alloc(Kel);
  float* Qp   = alloc(KI);
  float* Kp   = alloc(KI);
  float* Vp   = alloc(KI);
  float* ao   = alloc(KI);
  float* aout = alloc(Kel);
  float* mlp1 = alloc((size_t)B * NQ * MLPD);

  auto gemm = [&](const float* A_, const Lin& W_, float* C_, int M, int N, int KD, bool relu) {
    dim3 grid(N / 64, M / 64);
    if (relu) gemm_kernel<true ><<<grid, 128, 0, stream>>>(A_, W_.w, W_.b, C_, M, N, KD);
    else      gemm_kernel<false><<<grid, 128, 0, stream>>>(A_, W_.w, W_.b, C_, M, N, KD);
  };
  auto add = [&](const float* a, const float* b, float* c, size_t n) {
    add_kernel<<<(unsigned)((n + 255) / 256), 256, 0, stream>>>(a, b, c, (int)n);
  };
  auto lnorm = [&](const float* x, const float* res, const LNP& P, float* outp, int rows) {
    layernorm_kernel<<<rows, 256, 0, stream>>>(x, res, P.g, P.b, outp);
  };
  auto flash = [&](const float* Qb, const float* Kb, const float* Vb, const int* msk,
                   float* Ob, int Nq, int Nk, int width, int dh) {
    int blocks = (B * NH * (Nq / 16)) / 4;   // 4 waves per block, exact division
    float sc = (dh == 32) ? 0.1767766953f : 0.25f;   // 1/sqrt(d_head)
    if (dh == 32)
      flash_attn_kernel<32, false><<<blocks, 128, 0, stream>>>(Qb, Kb, Vb, nullptr, Ob, Nq, Nk, width, sc);
    else if (msk)
      flash_attn_kernel<16, true ><<<blocks, 128, 0, stream>>>(Qb, Kb, Vb, msk, Ob, Nq, Nk, width, sc);
    else
      flash_attn_kernel<16, false><<<blocks, 128, 0, stream>>>(Qb, Kb, Vb, nullptr, Ob, Nq, Nk, width, sc);
  };

  // ---- init state ----
  hipMemcpyAsync(qry,  seg, Qel * sizeof(float), hipMemcpyDeviceToDevice, stream);
  hipMemcpyAsync(keys, vlm, Kel * sizeof(float), hipMemcpyDeviceToDevice, stream);
  pe_kernel<<<(unsigned)((Kel + 255) / 256), 256, 0, stream>>>(pos, PE, (int)Kel);

  for (int i = 0; i < 2; ++i) {
    const LayerP& lp = L[i];
    // ---- self attention (layer0: q=k=v=queries, no residual) ----
    const float* qin = qry;
    if (i != 0) { add(qry, seg, qtmp, Qel); qin = qtmp; }
    gemm(qin, lp.self_attn.q, Qp, B * NQ, ED, ED, false);
    gemm(qin, lp.self_attn.k, Kp, B * NQ, ED, ED, false);
    gemm(qry, lp.self_attn.v, Vp, B * NQ, ED, ED, false);
    flash(Qp, Kp, Vp, nullptr, ao, NQ, NQ, ED, 32);
    gemm(ao, lp.self_attn.o, aout, B * NQ, ED, ED, false);
    lnorm(aout, (i == 0) ? nullptr : qry, lp.n1, qry, B * NQ);
    // ---- cross attention: tokens attend image (masked) ----
    add(qry, seg, qtmp, Qel);
    add(keys, PE, ktmp, Kel);
    gemm(qtmp, lp.cross_t2i.q, Qp, B * NQ, IDIM, ED, false);
    gemm(ktmp, lp.cross_t2i.k, Kp, B * KK, IDIM, ED, false);
    gemm(keys, lp.cross_t2i.v, Vp, B * KK, IDIM, ED, false);
    flash(Qp, Kp, Vp, amask, ao, NQ, KK, IDIM, 16);
    gemm(ao, lp.cross_t2i.o, aout, B * NQ, ED, IDIM, false);
    lnorm(aout, qry, lp.n2, qry, B * NQ);
    // ---- MLP ----
    gemm(qry, lp.lin1, mlp1, B * NQ, MLPD, ED, true);
    gemm(mlp1, lp.lin2, aout, B * NQ, ED, MLPD, false);
    lnorm(aout, qry, lp.n3, qry, B * NQ);
    // ---- cross attention: image attends tokens ----
    add(qry, seg, qtmp, Qel);
    add(keys, PE, ktmp, Kel);
    gemm(ktmp, lp.cross_i2t.q, Qp, B * KK, IDIM, ED, false);   // q-side = keys+pe
    gemm(qtmp, lp.cross_i2t.k, Kp, B * NQ, IDIM, ED, false);
    gemm(qry,  lp.cross_i2t.v, Vp, B * NQ, IDIM, ED, false);
    flash(Qp, Kp, Vp, nullptr, ao, KK, NQ, IDIM, 16);
    gemm(ao, lp.cross_i2t.o, aout, B * KK, ED, IDIM, false);
    lnorm(aout, keys, lp.n4, keys, B * KK);
  }

  // ---- final masked cross attention + final LN ----
  add(qry, seg, qtmp, Qel);
  add(keys, PE, ktmp, Kel);
  gemm(qtmp, finalA.q, Qp, B * NQ, IDIM, ED, false);
  gemm(ktmp, finalA.k, Kp, B * KK, IDIM, ED, false);
  gemm(keys, finalA.v, Vp, B * KK, IDIM, ED, false);
  flash(Qp, Kp, Vp, amask, ao, NQ, KK, IDIM, 16);
  gemm(ao, finalA.o, aout, B * NQ, ED, IDIM, false);
  lnorm(aout, qry, nF, qry, B * NQ);

  float* outp = (float*)d_out;
  hipMemcpyAsync(outp,       qry,  Qel * sizeof(float), hipMemcpyDeviceToDevice, stream);
  hipMemcpyAsync(outp + Qel, keys, Kel * sizeof(float), hipMemcpyDeviceToDevice, stream);
}
